// UAG_RNN_8Neigh_15908558864729
// MI455X (gfx1250) — compile-verified, matrix operand order checked by reading, not run on hardware
//
#include <hip/hip_runtime.h>

// ---------------------------------------------------------------------------
// UAG-RNN 8-neighbour spatial RNN for MI455X (gfx1250, wave32, WMMA)
// All channel-mixing 1x1 convs run as bf16 WMMA (v_wmma_f32_16x16x32_bf16)
// with f32 accumulation; recurrent state lives in LDS.
// ---------------------------------------------------------------------------

typedef __attribute__((ext_vector_type(16))) __bf16 v16bf;
typedef __attribute__((ext_vector_type(8)))  __bf16 v8bf;
typedef __attribute__((ext_vector_type(8)))  float  v8f;
typedef __attribute__((ext_vector_type(4)))  float  v4f;

#define CD   128   // channels
#define BB   8     // batch
#define HH   128   // height
#define WW   128   // width
#define NTOT (BB * CD * HH * WW)   // 16,777,216 elements

// ---- bf16 helpers ---------------------------------------------------------
__device__ __forceinline__ unsigned short f2bf_u(float f) {
  union { float f; unsigned u; } v; v.f = f;
  unsigned r = v.u + 0x7FFFu + ((v.u >> 16) & 1u);   // round-to-nearest-even
  return (unsigned short)(r >> 16);
}
__device__ __forceinline__ float bf2f(unsigned short s) {
  union { unsigned u; float f; } v; v.u = ((unsigned)s) << 16; return v.f;
}
__device__ __forceinline__ uint4 pack_bf8(v8f a) {
  uint4 o;
  o.x = (unsigned)f2bf_u(a[0]) | ((unsigned)f2bf_u(a[1]) << 16);
  o.y = (unsigned)f2bf_u(a[2]) | ((unsigned)f2bf_u(a[3]) << 16);
  o.z = (unsigned)f2bf_u(a[4]) | ((unsigned)f2bf_u(a[5]) << 16);
  o.w = (unsigned)f2bf_u(a[6]) | ((unsigned)f2bf_u(a[7]) << 16);
  return o;
}

// Load one 16-element bf16 fragment per lane, following the ISA 16-bit A/B
// striping: this lane's row, K-elements [ko, ko+8) and [ko+16, ko+24).
// 'row' points at the 128-channel contiguous row for this lane's M (or N),
// 'ko' = kchunk*32 + (lane&16 ? 8 : 0).
__device__ __forceinline__ v16bf load_frag(const unsigned short* row, int ko) {
  v8bf lo = *(const v8bf*)((const __bf16*)(row + ko));
  v8bf hi = *(const v8bf*)((const __bf16*)(row + ko + 16));
  v16bf r;
#pragma unroll
  for (int i = 0; i < 8; ++i) { r[i] = lo[i]; r[i + 8] = hi[i]; }
  return r;
}

#define WMMA_BF16(A, Bm, Cacc) \
  __builtin_amdgcn_wmma_f32_16x16x32_bf16(false, (A), false, (Bm), (short)0, (Cacc), false, false)

// ---- utility kernels ------------------------------------------------------
__global__ __launch_bounds__(256) void k_zero(float* out, int n) {
  for (int i = blockIdx.x * blockDim.x + threadIdx.x; i < n;
       i += gridDim.x * blockDim.x)
    out[i] = 0.f;
}

struct WPtrs { const float* p[16]; };

__global__ __launch_bounds__(256) void k_cvt_w(WPtrs wp, unsigned short* Wbf) {
  int i = blockIdx.x * blockDim.x + threadIdx.x;   // 16 * 16384 threads
  int wi = i >> 14;
  int j  = i & 16383;
  Wbf[i] = f2bf_u(wp.p[wi][j]);
}

// x (B,C,H,W) f32  ->  xT (H,B,W,C) bf16   (channels contiguous per position)
__global__ __launch_bounds__(256) void k_cvt_x(const float* __restrict__ x,
                                               unsigned short* __restrict__ xT) {
  for (int i = blockIdx.x * blockDim.x + threadIdx.x; i < NTOT;
       i += gridDim.x * blockDim.x) {
    int c = i & 127;
    int w = (i >> 7) & 127;
    int b = (i >> 14) & 7;
    int h = i >> 17;
    xT[i] = f2bf_u(x[(((size_t)b * CD + c) * HH + h) * WW + w]);
  }
}

// ---- row sweep (south: dir=+1, north: dir=-1) ----------------------------
// 64 blocks x 256 threads. Block owns 16 (b,w) positions; recurrence over h.
// dstT layout: (w, b, h, c) bf16  -> B-matrix layout for the column sweeps.
__global__ __launch_bounds__(256) void k_row_sweep(
    const unsigned short* __restrict__ xT, unsigned short* __restrict__ dstT,
    const unsigned short* __restrict__ Wbf, int wi_x, int wi_p,
    const float* __restrict__ bx, const float* __restrict__ bp, int dir) {
  __shared__ unsigned short prevS[2][16 * CD];   // 8 KB, double buffered

  const int tid   = threadIdx.x;
  const int wave  = tid >> 5;          // M-tile: channels [wave*16, wave*16+16)
  const int lane  = tid & 31;
  const int nlane = lane & 15;         // position within the 16-wide N tile
  const int khalf = (lane & 16) ? 8 : 0;
  const int p0 = blockIdx.x * 16;
  const int b  = p0 >> 7;
  const int w0 = p0 & 127;
  const int c0 = wave * 16 + ((lane >> 4) << 3);   // C-layout channel base

  // A fragments (weights, row-major CxC): W_x applied to x row, W_p to prev.
  const unsigned short* arow1 = Wbf + wi_x * 16384 + (wave * 16 + nlane) * CD;
  const unsigned short* arow2 = Wbf + wi_p * 16384 + (wave * 16 + nlane) * CD;
  v16bf A1[4], A2[4];
#pragma unroll
  for (int k = 0; k < 4; ++k) {
    A1[k] = load_frag(arow1, k * 32 + khalf);
    A2[k] = load_frag(arow2, k * 32 + khalf);
  }
  v8f bias;
#pragma unroll
  for (int r = 0; r < 8; ++r) bias[r] = bx[c0 + r] + bp[c0 + r];

  // init row: south -> x row 0 untouched; north -> relu(x row 127)
  const int h0 = (dir > 0) ? 0 : HH - 1;
  for (int i = tid; i < 16 * CD; i += blockDim.x) {
    int n = i >> 7, c = i & 127;
    float v = bf2f(xT[(((size_t)h0 * BB + b) * WW + (w0 + n)) * CD + c]);
    if (dir < 0) v = fmaxf(v, 0.f);
    unsigned short us = f2bf_u(v);
    prevS[0][i] = us;
    dstT[(((size_t)(w0 + n) * BB + b) * HH + h0) * CD + c] = us;
  }
  __syncthreads();

  int cb = 0;
  for (int s = 1; s < HH; ++s) {
    const int h = (dir > 0) ? s : (HH - 1 - s);
    const unsigned short* xrow =
        xT + (((size_t)h * BB + b) * WW + (w0 + nlane)) * CD;
    const unsigned short* prow = prevS[cb] + nlane * CD;
    v8f acc = bias;
#pragma unroll
    for (int k = 0; k < 4; ++k) {
      acc = WMMA_BF16(A1[k], load_frag(xrow, k * 32 + khalf), acc);
      acc = WMMA_BF16(A2[k], load_frag(prow, k * 32 + khalf), acc);
    }
#pragma unroll
    for (int r = 0; r < 8; ++r) acc[r] = fmaxf(acc[r], 0.f);
    uint4 pk = pack_bf8(acc);
    *(uint4*)(prevS[cb ^ 1] + nlane * CD + c0) = pk;
    *(uint4*)(dstT + (((size_t)(w0 + nlane) * BB + b) * HH + h) * CD + c0) = pk;
    __syncthreads();
    cb ^= 1;
  }
}

// ---- column sweep ---------------------------------------------------------
// 8 blocks (one per batch) x 256 threads. Serial over 128 columns; the
// height-shifted diagonal term couples h positions, so the whole 128-high
// column state stays in this block's LDS.
//   step: t = Wd@prev + bd (phase A, into LDS f32 buffer)
//         new = relu(Wc@cur + bc + Wp@prev + bp + gamma * t[h -/+ 1])
__global__ __launch_bounds__(256) void k_col_sweep(
    const unsigned short* __restrict__ srcT, const unsigned short* __restrict__ Wbf,
    int wi_d, int wi_c, int wi_p,
    const float* __restrict__ bd, const float* __restrict__ bc,
    const float* __restrict__ bp, const float* __restrict__ gptr,
    int reverse, int shiftDir, float* __restrict__ out) {
  extern __shared__ unsigned char smem[];
  unsigned short* prev0 = (unsigned short*)smem;            // 32 KB
  unsigned short* prev1 = (unsigned short*)(smem + 32768);  // 32 KB
  float* tbuf = (float*)(smem + 65536);                     // 64 KB

  const int tid   = threadIdx.x;
  const int wave  = tid >> 5;
  const int lane  = tid & 31;
  const int nlane = lane & 15;
  const int khalf = (lane & 16) ? 8 : 0;
  const int b  = blockIdx.x;
  const int c0 = wave * 16 + ((lane >> 4) << 3);
  const float gamma = gptr[0];

  v16bf Ad[4], Ac[4], Ap[4];
  {
    const unsigned short* rd = Wbf + wi_d * 16384 + (wave * 16 + nlane) * CD;
    const unsigned short* rc = Wbf + wi_c * 16384 + (wave * 16 + nlane) * CD;
    const unsigned short* rp = Wbf + wi_p * 16384 + (wave * 16 + nlane) * CD;
#pragma unroll
    for (int k = 0; k < 4; ++k) {
      Ad[k] = load_frag(rd, k * 32 + khalf);
      Ac[k] = load_frag(rc, k * 32 + khalf);
      Ap[k] = load_frag(rp, k * 32 + khalf);
    }
  }
  v8f biasD, biasCP;
#pragma unroll
  for (int r = 0; r < 8; ++r) {
    biasD[r]  = bd[c0 + r];
    biasCP[r] = bc[c0 + r] + bp[c0 + r];
  }

  // init column: relu only
  {
    int w = reverse ? (WW - 1) : 0;
    for (int i = tid; i < HH * CD; i += blockDim.x) {
      int h = i >> 7, c = i & 127;
      float v = fmaxf(bf2f(srcT[(((size_t)w * BB + b) * HH + h) * CD + c]), 0.f);
      prev0[i] = f2bf_u(v);
      out[(((size_t)b * CD + c) * HH + h) * WW + w] += v;
    }
  }
  __syncthreads();

  unsigned short* pv[2] = { prev0, prev1 };
  int cb = 0;
  for (int j = 1; j < WW; ++j) {
    const int w = reverse ? (WW - 1 - j) : j;

    // Phase A: diagonal pre-conv t = Wd@prev + bd for the whole column
#pragma unroll 2
    for (int nt = 0; nt < 8; ++nt) {
      const int h = nt * 16 + nlane;
      const unsigned short* prow = pv[cb] + h * CD;
      v8f t = biasD;
#pragma unroll
      for (int k = 0; k < 4; ++k)
        t = WMMA_BF16(Ad[k], load_frag(prow, k * 32 + khalf), t);
      float* tp = tbuf + h * CD + c0;
      *(v4f*)tp       = (v4f){t[0], t[1], t[2], t[3]};
      *(v4f*)(tp + 4) = (v4f){t[4], t[5], t[6], t[7]};
    }
    __syncthreads();

    // Phase B: new = relu(Wc@cur + Wp@prev + bias + gamma * shift(t))
#pragma unroll 2
    for (int nt = 0; nt < 8; ++nt) {
      const int h = nt * 16 + nlane;
      const unsigned short* crow = srcT + (((size_t)w * BB + b) * HH + h) * CD;
      const unsigned short* prow = pv[cb] + h * CD;
      v8f acc = biasCP;
#pragma unroll
      for (int k = 0; k < 4; ++k) {
        acc = WMMA_BF16(Ac[k], load_frag(crow, k * 32 + khalf), acc);
        acc = WMMA_BF16(Ap[k], load_frag(prow, k * 32 + khalf), acc);
      }
      const int hs = h + shiftDir;
      if (hs >= 0 && hs < HH) {
        const float* tp = tbuf + hs * CD + c0;
#pragma unroll
        for (int r = 0; r < 8; ++r) acc[r] += gamma * tp[r];
      }
#pragma unroll
      for (int r = 0; r < 8; ++r) acc[r] = fmaxf(acc[r], 0.f);
      *(uint4*)(pv[cb ^ 1] + h * CD + c0) = pack_bf8(acc);
      float* op = out + (((size_t)b * CD + c0) * HH + h) * WW + w;
#pragma unroll
      for (int r = 0; r < 8; ++r) op[(size_t)r * HH * WW] += acc[r];
    }
    __syncthreads();
    cb ^= 1;
  }
}

// ---------------------------------------------------------------------------
extern "C" void kernel_launch(void* const* d_in, const int* in_sizes, int n_in,
                              void* d_out, int out_size, void* d_ws, size_t ws_size,
                              hipStream_t stream) {
  (void)in_sizes; (void)n_in; (void)out_size; (void)ws_size;

  const float* x = (const float*)d_in[0];
  const float* W[16];
  const float* bvec[16];
  const float* g[4];
  for (int i = 0; i < 16; ++i) { W[i] = (const float*)d_in[1 + i]; }
  for (int i = 0; i < 16; ++i) { bvec[i] = (const float*)d_in[17 + i]; }
  for (int i = 0; i < 4;  ++i) { g[i] = (const float*)d_in[33 + i]; }
  float* out = (float*)d_out;

  // workspace layout (bf16 buffers)
  unsigned char* ws = (unsigned char*)d_ws;
  unsigned short* Wbf = (unsigned short*)ws;                       // 512 KB
  unsigned short* xT  = (unsigned short*)(ws + 524288);            // 32 MB (H,B,W,C)
  unsigned short* hsT = (unsigned short*)(ws + 524288 + 33554432); // 32 MB (W,B,H,C)
  unsigned short* hnT = (unsigned short*)(ws + 524288 + 2u * 33554432); // 32 MB

  k_zero<<<4096, 256, 0, stream>>>(out, NTOT);

  WPtrs wp;
  for (int i = 0; i < 16; ++i) wp.p[i] = W[i];
  k_cvt_w<<<1024, 256, 0, stream>>>(wp, Wbf);
  k_cvt_x<<<8192, 256, 0, stream>>>(x, xT);

  // south sweep (W1 on x, W2 on prev) and north sweep (W9, W10)
  k_row_sweep<<<64, 256, 0, stream>>>(xT, hsT, Wbf, 0, 1, bvec[0], bvec[1], +1);
  k_row_sweep<<<64, 256, 0, stream>>>(xT, hnT, Wbf, 8, 9, bvec[8], bvec[9], -1);

  const size_t col_lds = 131072;   // 2 x 32KB state + 64KB t-buffer
  // hse: Wd=W3 Wc=W4 Wp=W5, g1, shift_down (h-1), forward
  k_col_sweep<<<8, 256, col_lds, stream>>>(hsT, Wbf, 2, 3, 4,
      bvec[2], bvec[3], bvec[4], g[0], 0, -1, out);
  // hsw: Wd=W6 Wc=W7 Wp=W8, g2, shift_down, reverse
  k_col_sweep<<<8, 256, col_lds, stream>>>(hsT, Wbf, 5, 6, 7,
      bvec[5], bvec[6], bvec[7], g[1], 1, -1, out);
  // hne: Wd=W11 Wc=W12 Wp=W13, g3, shift_up (h+1), forward
  k_col_sweep<<<8, 256, col_lds, stream>>>(hnT, Wbf, 10, 11, 12,
      bvec[10], bvec[11], bvec[12], g[2], 0, +1, out);
  // hnw: Wd=W14 Wc=W15 Wp=W16, g4, shift_up, reverse
  k_col_sweep<<<8, 256, col_lds, stream>>>(hnT, Wbf, 13, 14, 15,
      bvec[13], bvec[14], bvec[15], g[3], 1, +1, out);
}
